// NCPQNetwork_23106924052949
// MI455X (gfx1250) — matrix operand order repaired and checked by reading, not compiled, not run on hardware
//
#include <hip/hip_runtime.h>
#include <hip/hip_bf16.h>

// ---------------- problem constants (from reference) ----------------
#define OBS     64
#define UNITS   114          // ACT_DIM + COMMAND + INTER
#define ACT     18
#define QH      114          // Q_HIDDEN = max(32, UNITS)
#define BATCH   2048
#define TSTEPS  64
#define UNFOLDS 6
#define EPS     1e-8f
#define LOG2E   1.4426950408889634f

typedef __attribute__((ext_vector_type(16))) _Float16 v16h;
typedef __attribute__((ext_vector_type(8)))  float    v8f;

// =====================================================================
// Kernel 1: fuse per-edge constants.
//   rec[i*U+j] = { ww, ww*erev, sigma*log2e, mu*sigma*log2e }
//   sen[s*U+j] = same for sensory synapses
// sigmoid((v-mu)*sigma) == rcp(1 + exp2(musig2 - v*sig2))
// =====================================================================
__global__ void ltc_precompute(const float* __restrict__ w,    const float* __restrict__ mu,
                               const float* __restrict__ sigma,const float* __restrict__ erev,
                               const float* __restrict__ mask,
                               const float* __restrict__ sw_,  const float* __restrict__ smu,
                               const float* __restrict__ ssig, const float* __restrict__ serev,
                               const float* __restrict__ smask,
                               float4* __restrict__ rec, float4* __restrict__ sen) {
  int idx = blockIdx.x * blockDim.x + threadIdx.x;
  if (idx < UNITS * UNITS) {
    float ww = w[idx] * mask[idx];
    rec[idx] = make_float4(ww, ww * erev[idx],
                           sigma[idx] * LOG2E, mu[idx] * sigma[idx] * LOG2E);
  }
  if (idx < OBS * UNITS) {
    float swv = sw_[idx] * smask[idx];
    sen[idx] = make_float4(swv, swv * serev[idx],
                           ssig[idx] * LOG2E, smu[idx] * ssig[idx] * LOG2E);
  }
}

// =====================================================================
// Kernel 2: LTC recurrence. One block = 8 batch elements, 256 threads.
//   tid = (squad<<7) | j ; squad 0 -> batches 0..3, squad 1 -> 4..7
//   Recurrent weight table (203KB float4) lives in LDS for the whole
//   kernel; v is kept transposed [UNITS][8] so the inner loop is:
//     1x ds_load_b128 (weights) + 1x ds_load_b128 broadcast (v)
//     -> 4 batches x (fma, v_exp_f32, add, v_rcp_f32, fma, fma)
// =====================================================================
__global__ __launch_bounds__(256)
void ltc_main(const float* __restrict__ obs,     const float* __restrict__ hidden,
              const float* __restrict__ input_w, const float* __restrict__ input_b,
              const float4* __restrict__ recG,   const float4* __restrict__ senG,
              const float* __restrict__ gleak,   const float* __restrict__ vleak,
              const float* __restrict__ cm,      float* __restrict__ h_out) {
  extern __shared__ char smem[];
  float4* recS = (float4*)smem;                                   // 114*114*16 = 207936 B
  float*  vS   = (float*)(smem + UNITS * UNITS * sizeof(float4)); // [UNITS][8] = 3648 B
  float*  xiS  = vS + UNITS * 8;                                  // [OBS][8]   = 2048 B

  const int tid   = threadIdx.x;
  const int j     = tid & 127;          // post-neuron index (active if < 114)
  const int sb    = (tid >> 7) * 4;     // batch sub-slot: 0 or 4
  const int bbase = blockIdx.x * 8;

  // stage recurrent weights (shared across all 8 batches, all T, all unfolds)
  for (int idx = tid; idx < UNITS * UNITS; idx += 256) recS[idx] = recG[idx];

  float cmt = 0.f, gl = 0.f, glvl = 0.f;
  if (j < UNITS) {
    cmt  = cm[j] * (float)UNFOLDS;      // cm_t = cm * ode_unfolds
    gl   = gleak[j];
    glvl = gl * vleak[j];
    if (sb == 0) {                       // squad 0 initializes v for all 8 batches
      #pragma unroll
      for (int b = 0; b < 8; ++b)
        vS[j * 8 + b] = hidden[(size_t)(bbase + b) * UNITS + j];
    }
  }
  __syncthreads();

  for (int t = 0; t < TSTEPS; ++t) {
    // ---- stage xi = obs*input_w + input_b, transposed [s][b] ----
    for (int idx = tid; idx < 8 * OBS; idx += 256) {
      int b = idx >> 6, s = idx & 63;
      float x = obs[(size_t)(bbase + b) * (TSTEPS * OBS) + t * OBS + s];
      xiS[s * 8 + b] = x * input_w[s] + input_b[s];
    }
    __syncthreads();

    // ---- sensory pass -> per-timestep num/den bases ----
    float bnum[4], bden[4];
    #pragma unroll
    for (int k = 0; k < 4; ++k) { bnum[k] = glvl; bden[k] = cmt + gl; }
    if (j < UNITS) {
      #pragma unroll 2
      for (int s = 0; s < OBS; ++s) {
        float4 sw4 = senG[s * UNITS + j];              // L2-resident
        const float4 xv = *(const float4*)&xiS[s * 8 + sb];
        float xa[4] = {xv.x, xv.y, xv.z, xv.w};
        #pragma unroll
        for (int k = 0; k < 4; ++k) {
          float f = fmaf(-xa[k], sw4.z, sw4.w);        // (mu - xi)*sigma*log2e
          float r = __builtin_amdgcn_rcpf(1.0f + __builtin_amdgcn_exp2f(f));
          bden[k] = fmaf(sw4.x, r, bden[k]);
          bnum[k] = fmaf(sw4.y, r, bnum[k]);
        }
      }
    }

    // ---- ODE unfolds ----
    for (int u = 0; u < UNFOLDS; ++u) {
      float num[4], den[4];
      if (j < UNITS) {
        #pragma unroll
        for (int k = 0; k < 4; ++k) {
          num[k] = fmaf(cmt, vS[j * 8 + sb + k], bnum[k]);
          den[k] = bden[k];
        }
        #pragma unroll 2
        for (int i = 0; i < UNITS; ++i) {
          float4 rw = recS[i * UNITS + j];                       // ds_load_b128
          const float4 vv = *(const float4*)&vS[i * 8 + sb];     // broadcast b128
          float va[4] = {vv.x, vv.y, vv.z, vv.w};
          #pragma unroll
          for (int k = 0; k < 4; ++k) {
            float f = fmaf(-va[k], rw.z, rw.w);
            float r = __builtin_amdgcn_rcpf(1.0f + __builtin_amdgcn_exp2f(f));
            den[k] = fmaf(rw.x, r, den[k]);                      // += ww * sig
            num[k] = fmaf(rw.y, r, num[k]);                      // += ww*erev * sig
          }
        }
      }
      __syncthreads();   // all reads of v done before overwrite
      if (j < UNITS) {
        float4 vn;
        vn.x = num[0] * __builtin_amdgcn_rcpf(den[0] + EPS);
        vn.y = num[1] * __builtin_amdgcn_rcpf(den[1] + EPS);
        vn.z = num[2] * __builtin_amdgcn_rcpf(den[2] + EPS);
        vn.w = num[3] * __builtin_amdgcn_rcpf(den[3] + EPS);
        *(float4*)&vS[j * 8 + sb] = vn;                          // ds_store_b128
      }
      __syncthreads();
    }
  }

  // ---- write final hidden state ----
  if (j < UNITS) {
    #pragma unroll
    for (int k = 0; k < 4; ++k)
      h_out[(size_t)(bbase + sb + k) * UNITS + j] = vS[j * 8 + sb + k];
  }
}

// =====================================================================
// Kernel 3: Q-head via WMMA f32_16x16x32_f16, fragment-ready LDS.
//   All f16 operands are stored in LDS pre-permuted into the exact
//   per-lane WMMA register order [tile][kk][lane][16 halves], so each
//   fragment is ONE contiguous 32B read per lane (2x ds_load_b128)
//   instead of 16 scalar ds_load_u16.
//
//   A 16x32 f16 (contiguous k, same as B):  row = lane&15; hi = lane>>4
//     hi=0: K=e        (K in 0-15)
//     hi=1: K=e+16     (K in 16-31)
//     inverse (M,K): kb=K&31; hi=kb>>4; e=kb&15; lane=hi*16+M
//   B 32x16 f16: n = lane&15; K = kk*32 + (lane>>4)*16 + e
//     inverse (K,n): kk=K>>5; lane=((K>>4)&1)*16+(n&15); e=K&15
//   C/D: element r -> M = r + 8*(lane>=16), N = lane&15
// =====================================================================
__global__ __launch_bounds__(128)
void qhead(const float* __restrict__ h,    const float* __restrict__ q_w1,
           const float* __restrict__ q_b1, const float* __restrict__ q_w2,
           const float* __restrict__ q_b2, float* __restrict__ q_out) {
  extern __shared__ char smem[];
  _Float16* w1f = (_Float16*)smem;          // [8 nt][4 kk][32 lane][16] = 32768 B
  _Float16* w2f = w1f + 8 * 4 * 32 * 16;    // [2 nt][4 kk][32 lane][16] =  8192 B
  _Float16* hAf = w2f + 2 * 4 * 32 * 16;    // 4 waves x [4 kk][32][16]  = 16384 B
  _Float16* h1f = hAf + 4 * 4 * 32 * 16;    // 4 waves x [4 kk][32][16]  = 16384 B

  const int tid  = threadIdx.x;
  const int lane = tid & 31;
  const int wave = tid >> 5;
  const int row  = lane & 15;
  const int hi   = lane >> 4;
  const int bbase = blockIdx.x * 64 + wave * 16;

  // ---- stage W1 into B-fragment order (coalesced global reads) ----
  for (int idx = tid; idx < 128 * 128; idx += 128) {
    int K = idx >> 7, n = idx & 127;
    _Float16 val = (K < UNITS && n < QH) ? (_Float16)q_w1[K * QH + n] : (_Float16)0.f;
    int kk = K >> 5, bl = ((K >> 4) & 1) * 16 + (n & 15), e = K & 15, nt = n >> 4;
    w1f[(((nt * 4 + kk) * 32) + bl) * 16 + e] = val;
  }
  // ---- stage W2 into B-fragment order ----
  for (int idx = tid; idx < 128 * 32; idx += 128) {
    int K = idx >> 5, n = idx & 31;
    _Float16 val = (K < QH && n < ACT) ? (_Float16)q_w2[K * ACT + n] : (_Float16)0.f;
    int kk = K >> 5, bl = ((K >> 4) & 1) * 16 + (n & 15), e = K & 15, nt = n >> 4;
    w2f[(((nt * 4 + kk) * 32) + bl) * 16 + e] = val;
  }
  // ---- stage this wave's h tile into A-fragment order ----
  _Float16* myA  = hAf + wave * 4 * 32 * 16;
  _Float16* myH1 = h1f + wave * 4 * 32 * 16;
  for (int idx = lane; idx < 16 * 128; idx += 32) {
    int r = idx >> 7, K = idx & 127;
    _Float16 val = (K < UNITS) ? (_Float16)h[(size_t)(bbase + r) * UNITS + K]
                               : (_Float16)0.f;
    int kk = K >> 5, kb = K & 31;
    int ahi = kb >> 4, ae = kb & 15;
    myA[((kk * 32) + ahi * 16 + r) * 16 + ae] = val;
  }
  __syncthreads();

  // ---------- layer 1: hid1 = relu(h @ W1 + b1), K padded to 128 ----------
  #pragma unroll
  for (int nt = 0; nt < 8; ++nt) {
    v8f acc = {0.f, 0.f, 0.f, 0.f, 0.f, 0.f, 0.f, 0.f};
    #pragma unroll
    for (int kk = 0; kk < 4; ++kk) {
      v16h a = *(const v16h*)&myA[((kk * 32) + lane) * 16];
      v16h b = *(const v16h*)&w1f[(((nt * 4 + kk) * 32) + lane) * 16];
      acc = __builtin_amdgcn_wmma_f32_16x16x32_f16(false, a, false, b,
                                                   (short)0, acc, false, false);
    }
    // epilogue: bias + relu, scatter D straight into layer-2 A-fragment slots
    int N = nt * 16 + row;
    float bias = (N < QH) ? q_b1[N] : 0.f;
    int kk2 = N >> 5, kb2 = N & 31;
    int ahi2 = kb2 >> 4, ae2 = kb2 & 15;
    #pragma unroll
    for (int r = 0; r < 8; ++r) {           // D: M = r + 8*hi, N = nt*16+row
      int M = r + hi * 8;
      float v = acc[r] + bias;
      v = v > 0.f ? v : 0.f;                // pad cols (N>=114) stay exactly 0
      myH1[((kk2 * 32) + ahi2 * 16 + M) * 16 + ae2] = (_Float16)v;
    }
  }
  __syncthreads();

  // ---------- layer 2: q = hid1 @ W2 + b2 (N padded 18->32) ----------
  #pragma unroll
  for (int nt = 0; nt < 2; ++nt) {
    v8f acc = {0.f, 0.f, 0.f, 0.f, 0.f, 0.f, 0.f, 0.f};
    #pragma unroll
    for (int kk = 0; kk < 4; ++kk) {
      v16h a = *(const v16h*)&myH1[((kk * 32) + lane) * 16];
      v16h b = *(const v16h*)&w2f[(((nt * 4 + kk) * 32) + lane) * 16];
      acc = __builtin_amdgcn_wmma_f32_16x16x32_f16(false, a, false, b,
                                                   (short)0, acc, false, false);
    }
    int N = nt * 16 + row;
    if (N < ACT) {
      float bias = q_b2[N];
      #pragma unroll
      for (int r = 0; r < 8; ++r) {
        int M = r + hi * 8;
        q_out[(size_t)(bbase + M) * ACT + N] = acc[r] + bias;
      }
    }
  }
}

// =====================================================================
// Host-side launcher
// =====================================================================
extern "C" void kernel_launch(void* const* d_in, const int* in_sizes, int n_in,
                              void* d_out, int out_size, void* d_ws, size_t ws_size,
                              hipStream_t stream) {
  (void)in_sizes; (void)n_in; (void)out_size; (void)ws_size;

  const float* obs          = (const float*)d_in[0];
  const float* hidden       = (const float*)d_in[1];
  const float* input_w      = (const float*)d_in[2];
  const float* input_b      = (const float*)d_in[3];
  const float* sensory_w    = (const float*)d_in[4];
  const float* sensory_mu   = (const float*)d_in[5];
  const float* sensory_sig  = (const float*)d_in[6];
  const float* sensory_erev = (const float*)d_in[7];
  const float* sensory_mask = (const float*)d_in[8];
  const float* w            = (const float*)d_in[9];
  const float* mu           = (const float*)d_in[10];
  const float* sigma        = (const float*)d_in[11];
  const float* erev         = (const float*)d_in[12];
  const float* mask         = (const float*)d_in[13];
  const float* gleak        = (const float*)d_in[14];
  const float* vleak        = (const float*)d_in[15];
  const float* cm           = (const float*)d_in[16];
  const float* q_w1         = (const float*)d_in[17];
  const float* q_b1         = (const float*)d_in[18];
  const float* q_w2         = (const float*)d_in[19];
  const float* q_b2         = (const float*)d_in[20];

  float* q_out = (float*)d_out;                       // (2048, 18)
  float* h_out = q_out + (size_t)BATCH * ACT;         // (2048, 114)

  float4* rec = (float4*)d_ws;                        // 114*114*16 = 207936 B
  float4* sen = rec + UNITS * UNITS;                  //  64*114*16 = 116736 B

  // 1) fuse per-edge constants
  {
    int n = UNITS * UNITS;
    ltc_precompute<<<(n + 255) / 256, 256, 0, stream>>>(
        w, mu, sigma, erev, mask,
        sensory_w, sensory_mu, sensory_sig, sensory_erev, sensory_mask,
        rec, sen);
  }

  // 2) LTC recurrence (203KB weight table in LDS, >64KB dynamic LDS)
  {
    size_t lds = (size_t)UNITS * UNITS * sizeof(float4)  // recurrent weights
               + (size_t)UNITS * 8 * sizeof(float)       // v  [U][8]
               + (size_t)OBS * 8 * sizeof(float);        // xi [S][8]
    hipFuncSetAttribute((const void*)ltc_main,
                        hipFuncAttributeMaxDynamicSharedMemorySize, (int)lds);
    ltc_main<<<dim3(BATCH / 8), 256, lds, stream>>>(
        obs, hidden, input_w, input_b, rec, sen, gleak, vleak, cm, h_out);
  }

  // 3) Q-head GEMMs with WMMA
  {
    size_t lds = (size_t)(8 + 2 + 4 + 4) * 4 * 32 * 16 * sizeof(_Float16); // 73728 B
    hipFuncSetAttribute((const void*)qhead,
                        hipFuncAttributeMaxDynamicSharedMemorySize, (int)lds);
    qhead<<<dim3(BATCH / 64), 128, lds, stream>>>(h_out, q_w1, q_b1, q_w2, q_b2, q_out);
  }
}
